// DisentangledSelfAttention_26620207300896
// MI455X (gfx1250) — matrix operand order repaired and checked by reading, not compiled, not run on hardware
//
#include <hip/hip_runtime.h>
#include <hip/hip_bf16.h>
#include <cstdint>
#include <cstddef>

typedef __bf16 bf16;
typedef __attribute__((ext_vector_type(16))) __bf16 v16bf;
typedef __attribute__((ext_vector_type(8)))  float  v8f;

#define E_DIM 768
#define SSEQ  1024
#define KREL  512
#define NCHUNK 32
#define SCALE_F 0.07216878364870323f   // 1/sqrt(3*64)

union Frag { v16bf v; uint4 q[2]; };

__device__ __forceinline__ v8f wmma_bf16(v16bf a, v16bf b, v8f c) {
  return __builtin_amdgcn_wmma_f32_16x16x32_bf16(false, a, false, b, (short)0, c, false, false);
}

// compiler-level memory fence; HW keeps per-wave LDS ops in order
__device__ __forceinline__ void wave_fence() {
  __builtin_amdgcn_wave_barrier();
  asm volatile("" ::: "memory");
}

// CDNA5 async global->LDS copy (16B per lane), tracked by ASYNCcnt.
__device__ __forceinline__ void async_copy_b128(unsigned lds_off, const void* g) {
  asm volatile("global_load_async_to_lds_b128 %0, %1, off"
               :: "v"(lds_off), "v"((unsigned long long)(uintptr_t)g)
               : "memory");
}
__device__ __forceinline__ void wait_async0() {
  asm volatile("s_wait_asynccnt 0x0" ::: "memory");
}
__device__ __forceinline__ void wait_async1() {
  asm volatile("s_wait_asynccnt 0x1" ::: "memory");
}

// ---------------------------------------------------------------------------
// C[M,768] = A[M,768] @ W[768,768]^T + bias   (bf16 WMMA, f32 accumulate)
// Block tile 128Mx64N, 8 waves (4x2), wave tile 32x32.
// Software-pipelined: next K-chunk of A/W prefetched into registers.
// ---------------------------------------------------------------------------
template <bool A_BF16, bool OUT_BF16>
__global__ __launch_bounds__(256) void gemm_wmma_kernel(
    const void* __restrict__ Ain, const float* __restrict__ W,
    const float* __restrict__ bias, void* __restrict__ Cout, int M) {
  __shared__ __align__(16) bf16 As[128][40];  // 32-col K chunk, padded rows (80B)
  __shared__ __align__(16) bf16 Ws[64][40];

  const int tid = threadIdx.x;
  const int lane = tid & 31, w = tid >> 5;
  const int hl = lane >> 4, l16 = lane & 15;
  const int m0 = blockIdx.x * 128;
  const int n0 = blockIdx.y * 64;
  const int mi = w & 3, ni = w >> 2;

  // staging thread coords
  const int ar = tid >> 1, ac = (tid & 1) * 16;
  const int wn = tid >> 2, wc = (tid & 3) * 8;
  const bool a_ok = (m0 + ar) < M;
  const float* aF = (const float*)Ain + (size_t)(m0 + ar) * E_DIM + ac;
  const bf16*  aB = (const bf16*)Ain + (size_t)(m0 + ar) * E_DIM + ac;
  const float* wP = W + (size_t)(n0 + wn) * E_DIM + wc;

  float4 af[4] = {};
  uint4  ab[2] = {};
  float4 wf[2] = {};

  // prologue: prefetch K-chunk 0
  if (a_ok) {
    if (A_BF16) {
      const uint4* p = (const uint4*)aB;
      ab[0] = p[0]; ab[1] = p[1];
    } else {
      const float4* p = (const float4*)aF;
      af[0] = p[0]; af[1] = p[1]; af[2] = p[2]; af[3] = p[3];
    }
  }
  {
    const float4* p = (const float4*)wP;
    wf[0] = p[0]; wf[1] = p[1];
  }

  v8f acc[2][2] = {};

  for (int k0 = 0; k0 < E_DIM; k0 += 32) {
    __syncthreads();
    {  // store staged chunk (convert f32 -> bf16 if needed)
      bf16 tmp[16];
      if (a_ok) {
        if (A_BF16) {
          *(uint4*)(&tmp[0]) = ab[0];
          *(uint4*)(&tmp[8]) = ab[1];
        } else {
          const float* pf = (const float*)af;
#pragma unroll
          for (int e = 0; e < 16; ++e) tmp[e] = (bf16)pf[e];
        }
      } else {
#pragma unroll
        for (int e = 0; e < 16; ++e) tmp[e] = (bf16)0.0f;
      }
      *(uint4*)(&As[ar][ac])     = *(uint4*)(&tmp[0]);
      *(uint4*)(&As[ar][ac + 8]) = *(uint4*)(&tmp[8]);
      bf16 tw[8];
      const float* pw = (const float*)wf;
#pragma unroll
      for (int e = 0; e < 8; ++e) tw[e] = (bf16)pw[e];
      *(uint4*)(&Ws[wn][wc]) = *(uint4*)(&tw[0]);
    }
    if (k0 + 32 < E_DIM) {  // prefetch next chunk into registers
      if (a_ok) {
        if (A_BF16) {
          const uint4* p = (const uint4*)(aB + k0 + 32);
          ab[0] = p[0]; ab[1] = p[1];
        } else {
          const float4* p = (const float4*)(aF + k0 + 32);
          af[0] = p[0]; af[1] = p[1]; af[2] = p[2]; af[3] = p[3];
        }
      }
      const float4* p = (const float4*)(wP + k0 + 32);
      wf[0] = p[0]; wf[1] = p[1];
    }
    __syncthreads();

    Frag a[2], b[2];
#pragma unroll
    for (int t = 0; t < 2; ++t) {
      int r = mi * 32 + t * 16 + l16;
      a[t].q[0] = *(const uint4*)(&As[r][8 * hl]);
      a[t].q[1] = *(const uint4*)(&As[r][16 + 8 * hl]);
    }
#pragma unroll
    for (int t = 0; t < 2; ++t) {
      int n = ni * 32 + t * 16 + l16;
      b[t].q[0] = *(const uint4*)(&Ws[n][8 * hl]);
      b[t].q[1] = *(const uint4*)(&Ws[n][16 + 8 * hl]);
    }
#pragma unroll
    for (int i = 0; i < 2; ++i)
#pragma unroll
      for (int j = 0; j < 2; ++j) acc[i][j] = wmma_bf16(a[i].v, b[j].v, acc[i][j]);
  }

#pragma unroll
  for (int j = 0; j < 2; ++j) {
    int col = n0 + ni * 32 + j * 16 + l16;
    float bv = bias[col];
#pragma unroll
    for (int i = 0; i < 2; ++i)
#pragma unroll
      for (int v = 0; v < 8; ++v) {
        int row = m0 + mi * 32 + i * 16 + v + 8 * hl;
        if (row < M) {
          float val = acc[i][j][v] + bv;
          if (OUT_BF16)
            ((bf16*)Cout)[(size_t)row * E_DIM + col] = (bf16)val;
          else
            ((float*)Cout)[(size_t)row * E_DIM + col] = val;
        }
      }
  }
}

// ---------------------------------------------------------------------------
// Fused disentangled attention (flash-style online softmax).
// Grid: B*H*(S/128) blocks; 8 waves; wave owns 16 i-rows; j in chunks of 32.
// score = (Q.K^T + Q.KrBand^T[diag] + QrfBand.K^T[diag]) * SCALE
// K tiles double-buffered via async global->LDS DMA; V prefetched in regs.
// ---------------------------------------------------------------------------
__global__ __launch_bounds__(256) void attn_kernel(
    const bf16* __restrict__ qc, const bf16* __restrict__ kc,
    const bf16* __restrict__ vv, const bf16* __restrict__ qr,
    const bf16* __restrict__ kr, bf16* __restrict__ ctx) {
  __shared__ __align__(16) bf16  kc_s[2][32][72];   // K rows, padded (144B)
  __shared__ __align__(16) bf16  vt_s[2][64][40];   // V^T, padded (80B)
  __shared__ __align__(16) float scr[8][48][52];    // per-wave band/score scratch
  __shared__ __align__(16) bf16  p_s[8][16][40];    // per-wave P tile (80B rows)

  const int tid = threadIdx.x, lane = tid & 31, w = tid >> 5;
  const int hl = lane >> 4, l16 = lane & 15;
  const int blk = blockIdx.x;
  const int ib = blk & 7;
  const int h  = (blk >> 3) % 12;
  const int b  = blk / 96;
  const int iw = ib * 128 + w * 16;

  const size_t ho = (size_t)h * 64;
  const bf16* qbase = qc + (size_t)b * SSEQ * E_DIM + ho;
  const bf16* kbase = kc + (size_t)b * SSEQ * E_DIM + ho;
  const bf16* vbase = vv + (size_t)b * SSEQ * E_DIM + ho;
  const bf16* qrb = qr + ho;
  const bf16* krb = kr + ho;

  // staging coords: thread -> row jr, 8 cols at cst
  const int jr = tid >> 3, cst = (tid & 7) * 8;
  const bf16* krow0 = kbase + (size_t)jr * E_DIM + cst;
  const bf16* vrow0 = vbase + (size_t)jr * E_DIM + cst;

  // prologue: async K chunk 0, V chunk 0 into regs
  async_copy_b128((unsigned)(uintptr_t)&kc_s[0][jr][cst], krow0);
  uint4 vreg = *(const uint4*)(vrow0);

  // Q A-fragments (K=64 -> 2 frags), resident for whole kernel
  Frag qa[2];
  {
    const bf16* qrow = qbase + (size_t)(iw + l16) * E_DIM;
#pragma unroll
    for (int kc2 = 0; kc2 < 2; ++kc2) {
      qa[kc2].q[0] = *(const uint4*)(qrow + 32 * kc2 + 8 * hl);
      qa[kc2].q[1] = *(const uint4*)(qrow + 32 * kc2 + 16 + 8 * hl);
    }
  }

  v8f acc[4] = {};
  float mrun = -3.0e38f, lrun = 0.0f;

  for (int cidx = 0; cidx < NCHUNK; ++cidx) {
    const int slot = cidx & 1;
    const int j0 = cidx * 32;
    __syncthreads();  // (1) all waves done computing chunk cidx-1
    if (cidx + 1 < NCHUNK)
      async_copy_b128((unsigned)(uintptr_t)&kc_s[1 - slot][jr][cst],
                      krow0 + (size_t)(cidx + 1) * 32 * E_DIM);
    {  // V^T staging for current chunk (prefetched last iteration)
      bf16 vt[8];
      *(uint4*)vt = vreg;
#pragma unroll
      for (int e = 0; e < 8; ++e) vt_s[slot][cst + e][jr] = vt[e];
    }
    if (cidx + 1 < NCHUNK)
      vreg = *(const uint4*)(vrow0 + (size_t)(cidx + 1) * 32 * E_DIM);
    if (cidx + 1 < NCHUNK) wait_async1(); else wait_async0();
    __syncthreads();  // (2) kc_s[slot] + vt_s[slot] ready for all waves

    // ---- content-content: S = Q @ K^T ----
    Frag kb[2][2];
#pragma unroll
    for (int tj = 0; tj < 2; ++tj)
#pragma unroll
      for (int kc2 = 0; kc2 < 2; ++kc2) {
        kb[tj][kc2].q[0] =
            *(const uint4*)(&kc_s[slot][tj * 16 + l16][32 * kc2 + 8 * hl]);
        kb[tj][kc2].q[1] =
            *(const uint4*)(&kc_s[slot][tj * 16 + l16][32 * kc2 + 16 + 8 * hl]);
      }
    v8f S[2];
#pragma unroll
    for (int tj = 0; tj < 2; ++tj) {
      v8f s = {};
      s = wmma_bf16(qa[0].v, kb[tj][0].v, s);
      s = wmma_bf16(qa[1].v, kb[tj][1].v, s);
      S[tj] = s;
    }

    const int dlo = j0 - iw - 15;  // band: d = j - i in [dlo, dlo+46]

    // ---- content-position band: Q @ KrBand^T, rows=i, cols=band offset ----
#pragma unroll
    for (int tc = 0; tc < 3; ++tc) {
      int d = dlo + tc * 16 + l16;
      int rr = d + KREL;
      rr = rr < 0 ? 0 : (rr > 2 * KREL ? 2 * KREL : rr);
      const bf16* krow = krb + (size_t)rr * E_DIM;
      v8f cb = {};
#pragma unroll
      for (int kc2 = 0; kc2 < 2; ++kc2) {
        Frag bb;
        bb.q[0] = *(const uint4*)(krow + 32 * kc2 + 8 * hl);
        bb.q[1] = *(const uint4*)(krow + 32 * kc2 + 16 + 8 * hl);
        cb = wmma_bf16(qa[kc2].v, bb.v, cb);
      }
#pragma unroll
      for (int v = 0; v < 8; ++v) scr[w][v + 8 * hl][tc * 16 + l16] = cb[v];
    }
    wave_fence();
#pragma unroll
    for (int t = 0; t < 2; ++t)
#pragma unroll
      for (int v = 0; v < 8; ++v) {
        int m = v + 8 * hl, jj = t * 16 + l16;
        S[t][v] += scr[w][m][15 + jj - m];  // diagonal gather
      }
    wave_fence();

    // ---- position-content band: QrfBand @ K^T, rows=band offset, cols=j ----
    // qrf[rel] = q_r[clip(i-j)+K] -> row = clamp(K - d, 0, 2K)
#pragma unroll
    for (int u = 0; u < 4; ++u) {
      const int tc = (u == 0) ? 0 : (u == 3 ? 2 : 1);
      const int tj = (u >= 2) ? 1 : 0;
      int d = dlo + tc * 16 + l16;
      int rr = KREL - d;
      rr = rr < 0 ? 0 : (rr > 2 * KREL ? 2 * KREL : rr);
      const bf16* qrow = qrb + (size_t)rr * E_DIM;
      v8f cb = {};
#pragma unroll
      for (int kc2 = 0; kc2 < 2; ++kc2) {
        Frag aa;
        aa.q[0] = *(const uint4*)(qrow + 32 * kc2 + 8 * hl);
        aa.q[1] = *(const uint4*)(qrow + 32 * kc2 + 16 + 8 * hl);
        cb = wmma_bf16(aa.v, kb[tj][kc2].v, cb);
      }
#pragma unroll
      for (int v = 0; v < 8; ++v)
        scr[w][tc * 16 + v + 8 * hl][tj * 16 + l16] = cb[v];
    }
    wave_fence();
#pragma unroll
    for (int t = 0; t < 2; ++t)
#pragma unroll
      for (int v = 0; v < 8; ++v) {
        int m = v + 8 * hl, jj = t * 16 + l16;
        S[t][v] = (S[t][v] + scr[w][15 + jj - m][jj]) * SCALE_F;
      }
    wave_fence();

    // ---- store scores, per-row online softmax (lane <-> row layout) ----
#pragma unroll
    for (int t = 0; t < 2; ++t)
#pragma unroll
      for (int v = 0; v < 8; ++v) scr[w][v + 8 * hl][t * 16 + l16] = S[t][v];
    wave_fence();

    float4 xf[4];
    {
      const float4* rp = (const float4*)(&scr[w][l16][hl * 16]);
      xf[0] = rp[0]; xf[1] = rp[1]; xf[2] = rp[2]; xf[3] = rp[3];
    }
    float* x = (float*)xf;
    float mx = x[0];
#pragma unroll
    for (int e = 1; e < 16; ++e) mx = fmaxf(mx, x[e]);
    mx = fmaxf(mx, __shfl_xor(mx, 16, 32));
    float mnew = fmaxf(mrun, mx);
    float corr = __expf(mrun - mnew);
    float ssum = 0.0f;
    bf16 pb[16];
#pragma unroll
    for (int e = 0; e < 16; ++e) {
      float p = __expf(x[e] - mnew);
      ssum += p;
      pb[e] = (bf16)p;
    }
    ssum += __shfl_xor(ssum, 16, 32);
    lrun = lrun * corr + ssum;
    mrun = mnew;
    *(uint4*)(&p_s[w][l16][hl * 16])     = *(uint4*)(&pb[0]);
    *(uint4*)(&p_s[w][l16][hl * 16 + 8]) = *(uint4*)(&pb[8]);

    // rescale ctx accumulators (row factor broadcast via ds_bpermute)
#pragma unroll
    for (int v = 0; v < 8; ++v) {
      float cv = __int_as_float(
          __builtin_amdgcn_ds_bpermute((v + 8 * hl) * 4, __float_as_int(corr)));
#pragma unroll
      for (int nt = 0; nt < 4; ++nt) acc[nt][v] *= cv;
    }
    wave_fence();

    // ---- ctx += P @ V (K=32) ----
    Frag pa;
    pa.q[0] = *(const uint4*)(&p_s[w][l16][8 * hl]);
    pa.q[1] = *(const uint4*)(&p_s[w][l16][16 + 8 * hl]);
#pragma unroll
    for (int nt = 0; nt < 4; ++nt) {
      Frag vb;
      vb.q[0] = *(const uint4*)(&vt_s[slot][nt * 16 + l16][8 * hl]);
      vb.q[1] = *(const uint4*)(&vt_s[slot][nt * 16 + l16][16 + 8 * hl]);
      acc[nt] = wmma_bf16(pa.v, vb.v, acc[nt]);
    }
  }

  // finalize: divide by row sums, write ctx (bf16, [B,S,H,D] == [B,S,E])
  float rinv = 1.0f / lrun;
#pragma unroll
  for (int v = 0; v < 8; ++v) {
    float rv = __int_as_float(
        __builtin_amdgcn_ds_bpermute((v + 8 * hl) * 4, __float_as_int(rinv)));
    int row = iw + v + 8 * hl;
    bf16* crow = ctx + ((size_t)b * SSEQ + row) * E_DIM + ho;
#pragma unroll
    for (int nt = 0; nt < 4; ++nt) crow[nt * 16 + l16] = (bf16)(acc[nt][v] * rv);
  }
}

// ---------------------------------------------------------------------------
extern "C" void kernel_launch(void* const* d_in, const int* in_sizes, int n_in,
                              void* d_out, int out_size, void* d_ws,
                              size_t ws_size, hipStream_t stream) {
  (void)in_sizes; (void)n_in; (void)out_size; (void)ws_size;
  const float* x   = (const float*)d_in[0];
  const float* Wq  = (const float*)d_in[1];
  const float* bq  = (const float*)d_in[2];
  const float* Wk  = (const float*)d_in[3];
  const float* bk  = (const float*)d_in[4];
  const float* Wv  = (const float*)d_in[5];
  const float* bv  = (const float*)d_in[6];
  const float* Wpq = (const float*)d_in[7];
  const float* bpq = (const float*)d_in[8];
  const float* Wpk = (const float*)d_in[9];
  const float* bpk = (const float*)d_in[10];
  const float* rel = (const float*)d_in[11];
  const float* Wo  = (const float*)d_in[12];
  const float* bo  = (const float*)d_in[13];

  const size_t SZ_BSE = (size_t)4 * 1024 * 768 * sizeof(bf16);   // 6,291,456
  const size_t SZ_REL = (size_t)1025 * 768 * sizeof(bf16);       // 1,574,400
  char* ws = (char*)d_ws;
  bf16* qcw = (bf16*)(ws);
  bf16* kcw = (bf16*)(ws + SZ_BSE);
  bf16* vw  = (bf16*)(ws + 2 * SZ_BSE);
  bf16* cxw = (bf16*)(ws + 3 * SZ_BSE);
  bf16* qrw = (bf16*)(ws + 4 * SZ_BSE);
  bf16* krw = (bf16*)(ws + 4 * SZ_BSE + SZ_REL);

  dim3 blk(256);
  dim3 gproj(32, 12);  // M=4096
  dim3 grel(9, 12);    // M=1025
  gemm_wmma_kernel<false, true><<<gproj, blk, 0, stream>>>(x, Wq, bq, qcw, 4096);
  gemm_wmma_kernel<false, true><<<gproj, blk, 0, stream>>>(x, Wk, bk, kcw, 4096);
  gemm_wmma_kernel<false, true><<<gproj, blk, 0, stream>>>(x, Wv, bv, vw, 4096);
  gemm_wmma_kernel<false, true><<<grel, blk, 0, stream>>>(rel, Wpq, bpq, qrw, 1025);
  gemm_wmma_kernel<false, true><<<grel, blk, 0, stream>>>(rel, Wpk, bpk, krw, 1025);

  dim3 gattn(4 * 12 * 8);  // B*H*(S/128)
  attn_kernel<<<gattn, blk, 0, stream>>>(qcw, kcw, vw, qrw, krw, cxw);

  gemm_wmma_kernel<true, false><<<gproj, blk, 0, stream>>>(cxw, Wo, bo, d_out, 4096);
}
